// MultiHeadSelfAttention_27410481283257
// MI455X (gfx1250) — matrix operand order repaired and checked by reading, not compiled, not run on hardware
//
#include <hip/hip_runtime.h>
#include <hip/hip_bf16.h>
#include <math.h>

typedef __bf16 bf16_t;
typedef __attribute__((ext_vector_type(16))) __bf16 v16bf;
typedef __attribute__((ext_vector_type(8)))  __bf16 v8bf;
typedef __attribute__((ext_vector_type(4)))  __bf16 v4bf;
typedef __attribute__((ext_vector_type(8)))  float  v8f;
typedef __attribute__((ext_vector_type(4)))  float  v4f;

#define D_MODEL   1024
#define NUM_HEADS 16
#define DK        64
#define BATCH     4
#define SEQ       2048
#define ROWS      (BATCH * SEQ)   // 8192

union AB16 { v16bf v; v8bf h[2]; };

__device__ __forceinline__ v8f wmma_bf16(v16bf a, v16bf b, v8f c) {
  return __builtin_amdgcn_wmma_f32_16x16x32_bf16(false, a, false, b, (short)0, c, false, false);
}

// ---------------------------------------------------------------- converts
__global__ void f32_to_bf16_kernel(const float* __restrict__ src,
                                   bf16_t* __restrict__ dst, int n4) {
  int i = blockIdx.x * blockDim.x + threadIdx.x;
  if (i < n4) {
    v4f f = *(const v4f*)(src + 4 * (size_t)i);
    v4bf o;
    o[0] = (bf16_t)f[0]; o[1] = (bf16_t)f[1];
    o[2] = (bf16_t)f[2]; o[3] = (bf16_t)f[3];
    *(v4bf*)(dst + 4 * (size_t)i) = o;
  }
}

// ---------------------------------------------------------------- GEMM: Y = A * W^T
// One wave computes a 64x64 tile: 4x4 C fragments, 16 WMMAs per K-step.
// MODE 0: RoPE epilogue, store bf16 [B,H,S,64]
// MODE 1: store bf16 transposed [B,H,64,S]   (for V)
// MODE 2: store fp32 [ROWS, 1024]            (final projection)
template <int MODE>
__global__ __launch_bounds__(32)
void gemm_proj(const bf16_t* __restrict__ A,
               const bf16_t* __restrict__ W,
               void* __restrict__ out) {
  const int lane = threadIdx.x & 31;
  const int hi = lane >> 4, lq = lane & 15;
  const int m0 = blockIdx.x * 64;
  const int n0 = blockIdx.y * 64;

  const bf16_t* arow[4];
  const bf16_t* brow[4];
#pragma unroll
  for (int i = 0; i < 4; ++i) {
    arow[i] = A + (size_t)(m0 + i * 16 + lq) * D_MODEL;
    brow[i] = W + (size_t)(n0 + i * 16 + lq) * D_MODEL + hi * 16;
  }

  v8f acc[4][4] = {};
  for (int kb = 0; kb < D_MODEL; kb += 32) {
    AB16 a[4];
#pragma unroll
    for (int mt = 0; mt < 4; ++mt) {
      a[mt].h[0] = *(const v8bf*)(arow[mt] + kb + hi * 16);
      a[mt].h[1] = *(const v8bf*)(arow[mt] + kb + hi * 16 + 8);
    }
#pragma unroll
    for (int nt = 0; nt < 4; ++nt) {
      AB16 b;
      b.v = *(const v16bf*)(brow[nt] + kb);
#pragma unroll
      for (int mt = 0; mt < 4; ++mt)
        acc[mt][nt] = wmma_bf16(a[mt].v, b.v, acc[mt][nt]);
    }
  }

  if (MODE == 2) {
    float* O = (float*)out;
#pragma unroll
    for (int mt = 0; mt < 4; ++mt)
#pragma unroll
      for (int nt = 0; nt < 4; ++nt)
#pragma unroll
        for (int v = 0; v < 8; ++v) {
          int m = m0 + mt * 16 + v + 8 * hi;
          O[(size_t)m * D_MODEL + n0 + nt * 16 + lq] = acc[mt][nt][v];
        }
  } else {
    bf16_t* O = (bf16_t*)out;
#pragma unroll
    for (int nt = 0; nt < 4; ++nt) {
      const int n = n0 + nt * 16 + lq;
      const int h = n >> 6;     // head
      const int d = n & 63;     // dim within head
      if (MODE == 0) {
        // inv_freq = 10000^(-(d&~1)/64) = exp(-(d>>1) * ln(1e4)/32)
        const float inv_freq = __expf(-(float)(d >> 1) * 0.28782313662425f);
#pragma unroll
        for (int mt = 0; mt < 4; ++mt)
#pragma unroll
          for (int v = 0; v < 8; ++v) {
            int m = m0 + mt * 16 + v + 8 * hi;
            int s = m & (SEQ - 1);
            int bb = m >> 11;
            float sn, cs;
            __sincosf((float)s * inv_freq, &sn, &cs);
            float self = acc[mt][nt][v];
            float part = __shfl_xor(self, 1, 32);  // pair partner (n^1 stays in half)
            float r = ((d & 1) == 0) ? (self * cs - part * sn)
                                     : (part * sn + self * cs);
            O[(((size_t)(bb * NUM_HEADS + h)) * SEQ + s) * DK + d] = (bf16_t)r;
          }
      } else {  // MODE 1: V transposed [B,H,64,S]
#pragma unroll
        for (int mt = 0; mt < 4; ++mt)
#pragma unroll
          for (int v = 0; v < 8; ++v) {
            int m = m0 + mt * 16 + v + 8 * hi;
            int s = m & (SEQ - 1);
            int bb = m >> 11;
            O[(((size_t)(bb * NUM_HEADS + h)) * DK + d) * SEQ + s] = (bf16_t)acc[mt][nt][v];
          }
      }
    }
  }
}

// ---------------------------------------------------------------- attention
// Qb,Kb: [B,H,S,64] bf16 (RoPE applied).  Vt: [B,H,64,S] bf16.
// Ob: [B,S,1024] bf16.  One wave per 16-query tile, streaming 32-key blocks.
// Row-sum of P rides along as a 5th WMMA against an all-ones B fragment.
__global__ __launch_bounds__(32)
void attn_kernel(const bf16_t* __restrict__ Qb,
                 const bf16_t* __restrict__ Kb,
                 const bf16_t* __restrict__ Vt,
                 bf16_t* __restrict__ Ob) {
  __shared__ __align__(32) bf16_t Pt[16 * 32];   // P tile [q][key], bf16

  const int lane = threadIdx.x & 31;
  const int hi = lane >> 4;
  const int lq = lane & 15;
  const int q0 = blockIdx.x * 16;
  const int bh = blockIdx.y;                       // b*H + h
  const int bb = bh / NUM_HEADS, h = bh % NUM_HEADS;

  const bf16_t* qb_ = Qb + (size_t)bh * SEQ * DK;
  const bf16_t* kb_ = Kb + (size_t)bh * SEQ * DK;
  const bf16_t* vb_ = Vt + (size_t)bh * DK * SEQ;

  // Q A-fragments: d 0..31 (qa0) and 32..63 (qa1)
  const bf16_t* qrow = qb_ + (size_t)(q0 + lq) * DK;
  AB16 qa0, qa1;
  qa0.h[0] = *(const v8bf*)(qrow + hi * 16);
  qa0.h[1] = *(const v8bf*)(qrow + hi * 16 + 8);
  qa1.h[0] = *(const v8bf*)(qrow + 32 + hi * 16);
  qa1.h[1] = *(const v8bf*)(qrow + 32 + hi * 16 + 8);

  v16bf ones;
#pragma unroll
  for (int i = 0; i < 16; ++i) ones[i] = (bf16_t)1.0f;

  float mrow[8];
  v8f o0 = {}, o1 = {}, o2 = {}, o3 = {}, o4 = {};  // o4 = running row-sum l
#pragma unroll
  for (int v = 0; v < 8; ++v) mrow[v] = -1e30f;

  auto process = [&](int k0, bool masked) {
    if (k0 + 32 < SEQ) {  // prefetch next key block (speculative)
      __builtin_prefetch(kb_ + (size_t)(k0 + 32 + lq) * DK, 0, 0);
      __builtin_prefetch(vb_ + (size_t)lq * SEQ + k0 + 32, 0, 0);
    }
    // scores: two 16-key sub-tiles, each K(d)=64 via two WMMAs
    v8f s0 = {}, s1 = {};
    {
      const bf16_t* kr0 = kb_ + (size_t)(k0 + lq) * DK + hi * 16;
      const bf16_t* kr1 = kb_ + (size_t)(k0 + 16 + lq) * DK + hi * 16;
      AB16 b0, b1;
      b0.v = *(const v16bf*)(kr0);
      b1.v = *(const v16bf*)(kr0 + 32);
      s0 = wmma_bf16(qa0.v, b0.v, s0);
      s0 = wmma_bf16(qa1.v, b1.v, s0);
      b0.v = *(const v16bf*)(kr1);
      b1.v = *(const v16bf*)(kr1 + 32);
      s1 = wmma_bf16(qa0.v, b0.v, s1);
      s1 = wmma_bf16(qa1.v, b1.v, s1);
    }

    // online softmax over 32 keys (row max via shuffles in the 16-lane half)
#pragma unroll
    for (int v = 0; v < 8; ++v) {
      float a0 = s0[v] * 0.125f;                   // 1/sqrt(64)
      float a1 = s1[v] * 0.125f;
      if (masked) {
        const int q = q0 + v + 8 * hi;
        if (k0 + lq      > q) a0 = -1e30f;
        if (k0 + 16 + lq > q) a1 = -1e30f;
      }
      float mx = fmaxf(a0, a1);
      mx = fmaxf(mx, __shfl_xor(mx, 1, 32));
      mx = fmaxf(mx, __shfl_xor(mx, 2, 32));
      mx = fmaxf(mx, __shfl_xor(mx, 4, 32));
      mx = fmaxf(mx, __shfl_xor(mx, 8, 32));
      const float mnew = fmaxf(mrow[v], mx);
      const float p0 = __expf(a0 - mnew);
      const float p1 = __expf(a1 - mnew);
      const float alpha = __expf(mrow[v] - mnew);
      mrow[v] = mnew;
      o0[v] *= alpha; o1[v] *= alpha; o2[v] *= alpha; o3[v] *= alpha; o4[v] *= alpha;
      // stage P (C-layout row = v + 8*hi) into LDS for A-fragment re-layout
      Pt[(v + 8 * hi) * 32 + lq]      = (bf16_t)p0;
      Pt[(v + 8 * hi) * 32 + 16 + lq] = (bf16_t)p1;
    }

    // read P back as a 16x32 A-fragment (same-wave LDS RAW, dscnt-ordered)
    AB16 pa;
    const bf16_t* prow = Pt + lq * 32;
    pa.h[0] = *(const v8bf*)(prow + hi * 16);
    pa.h[1] = *(const v8bf*)(prow + hi * 16 + 8);

    // O += P @ V for the 4 d-tiles; l += P @ ones
    AB16 vf;
    vf.v = *(const v16bf*)(vb_ + (size_t)(lq +  0) * SEQ + k0 + hi * 16);
    o0 = wmma_bf16(pa.v, vf.v, o0);
    vf.v = *(const v16bf*)(vb_ + (size_t)(lq + 16) * SEQ + k0 + hi * 16);
    o1 = wmma_bf16(pa.v, vf.v, o1);
    vf.v = *(const v16bf*)(vb_ + (size_t)(lq + 32) * SEQ + k0 + hi * 16);
    o2 = wmma_bf16(pa.v, vf.v, o2);
    vf.v = *(const v16bf*)(vb_ + (size_t)(lq + 48) * SEQ + k0 + hi * 16);
    o3 = wmma_bf16(pa.v, vf.v, o3);
    o4 = wmma_bf16(pa.v, ones, o4);
  };

  const int nkb   = (q0 + 16 + 31) >> 5;  // total 32-key blocks (causal extent)
  const int nfull = q0 >> 5;              // blocks fully below the diagonal
  for (int t = 0; t < nfull; ++t) process(t * 32, false);
  for (int t = nfull; t < nkb; ++t) process(t * 32, true);

  // epilogue: normalize by l (all 16 columns of o4 hold the row-sum)
#pragma unroll
  for (int v = 0; v < 8; ++v) {
    const int q = q0 + v + 8 * hi;
    const float inv = 1.0f / o4[v];
    bf16_t* orow = Ob + ((size_t)bb * SEQ + q) * D_MODEL + h * DK;
    orow[lq]      = (bf16_t)(o0[v] * inv);
    orow[lq + 16] = (bf16_t)(o1[v] * inv);
    orow[lq + 32] = (bf16_t)(o2[v] * inv);
    orow[lq + 48] = (bf16_t)(o3[v] * inv);
  }
}

// ---------------------------------------------------------------- launch
extern "C" void kernel_launch(void* const* d_in, const int* in_sizes, int n_in,
                              void* d_out, int out_size, void* d_ws, size_t ws_size,
                              hipStream_t stream) {
  const float* x  = (const float*)d_in[0];
  // d_in[1] = token_positions == arange(SEQ); RoPE uses s directly
  const float* Wq = (const float*)d_in[2];
  const float* Wk = (const float*)d_in[3];
  const float* Wv = (const float*)d_in[4];
  const float* Wo = (const float*)d_in[5];

  char* ws = (char*)d_ws;
  size_t off = 0;
  auto alloc = [&](size_t bytes) -> void* {
    void* p = ws + off;
    off += (bytes + 255) & ~(size_t)255;
    return p;
  };
  bf16_t* xb  = (bf16_t*)alloc((size_t)ROWS * D_MODEL * 2);
  bf16_t* wqb = (bf16_t*)alloc((size_t)D_MODEL * D_MODEL * 2);
  bf16_t* wkb = (bf16_t*)alloc((size_t)D_MODEL * D_MODEL * 2);
  bf16_t* wvb = (bf16_t*)alloc((size_t)D_MODEL * D_MODEL * 2);
  bf16_t* wob = (bf16_t*)alloc((size_t)D_MODEL * D_MODEL * 2);
  bf16_t* Qb  = (bf16_t*)alloc((size_t)ROWS * D_MODEL * 2);   // [B,H,S,64]
  bf16_t* Kb  = (bf16_t*)alloc((size_t)ROWS * D_MODEL * 2);   // [B,H,S,64]
  bf16_t* Vt  = (bf16_t*)alloc((size_t)ROWS * D_MODEL * 2);   // [B,H,64,S]
  bf16_t* Ob  = (bf16_t*)alloc((size_t)ROWS * D_MODEL * 2);   // [B,S,1024]

  const int nx4 = ROWS * D_MODEL / 4;
  const int nw4 = D_MODEL * D_MODEL / 4;
  f32_to_bf16_kernel<<<(nx4 + 255) / 256, 256, 0, stream>>>(x, xb, nx4);
  f32_to_bf16_kernel<<<(nw4 + 255) / 256, 256, 0, stream>>>(Wq, wqb, nw4);
  f32_to_bf16_kernel<<<(nw4 + 255) / 256, 256, 0, stream>>>(Wk, wkb, nw4);
  f32_to_bf16_kernel<<<(nw4 + 255) / 256, 256, 0, stream>>>(Wv, wvb, nw4);
  f32_to_bf16_kernel<<<(nw4 + 255) / 256, 256, 0, stream>>>(Wo, wob, nw4);

  dim3 ggrid(ROWS / 64, D_MODEL / 64);
  gemm_proj<0><<<ggrid, 32, 0, stream>>>(xb, wqb, Qb);   // Q + RoPE
  gemm_proj<0><<<ggrid, 32, 0, stream>>>(xb, wkb, Kb);   // K + RoPE
  gemm_proj<1><<<ggrid, 32, 0, stream>>>(xb, wvb, Vt);   // V transposed

  dim3 agrid(SEQ / 16, BATCH * NUM_HEADS);
  attn_kernel<<<agrid, 32, 0, stream>>>(Qb, Kb, Vt, Ob);

  gemm_proj<2><<<ggrid, 32, 0, stream>>>(Ob, wob, d_out); // output projection
}
